// RNN_24335284699494
// MI455X (gfx1250) — matrix operand order repaired and checked by reading, not compiled, not run on hardware
//
#include <hip/hip_runtime.h>

// ---------------------------------------------------------------------------
// RNN on MI455X (gfx1250): persistent-kernel recurrence with bf16 WMMA.
//   h_t = tanh(Wxh[x[:,t]] + h_{t-1} @ Whh_w^T + Whh_b)   (512 serial steps)
//   out = h_T @ Why_w^T + Why_b                            (64 x 32000)
// ---------------------------------------------------------------------------

typedef __bf16 bf16;
typedef __attribute__((ext_vector_type(16))) __bf16 v16bf;
typedef __attribute__((ext_vector_type(8)))  __bf16 v8bf;
typedef __attribute__((ext_vector_type(8)))  float  v8f;

#define HID   1024
#define BATCH 64
#define TSEQ  512
#define NOUT  32000

#define NWG     16      // persistent workgroups (16 x 64 cols = 1024)
#define THREADS 256     // 8 waves (wave32)

static __device__ __forceinline__ unsigned short f32_to_bf16_rne(float f) {
    unsigned u = __builtin_bit_cast(unsigned, f);
    unsigned r = u + 0x7FFFu + ((u >> 16) & 1u);
    return (unsigned short)(r >> 16);
}

// Branch-free tanh on the TRANS pipe: tanh(x) = 1 - 2/(exp(2x)+1).
// exp2f -> v_exp_f32, rcpf -> v_rcp_f32. Saturates correctly for |x| large.
static __device__ __forceinline__ float fast_tanh(float x) {
    float ex = __builtin_amdgcn_exp2f(x * 2.8853900817779268f);  // exp(2x)
    return 1.0f - 2.0f * __builtin_amdgcn_rcpf(ex + 1.0f);
}

union FragU {
    v16bf v;
    v8bf  h[2];
    unsigned short s[16];
};

// A fragment, 16-bit A 16x32 layout (contiguous K order, matching B):
// lane half h = lane>>4; VGPR0-7 hold K in [h*16, h*16+16).
static __device__ __forceinline__ v16bf load_frag_a(const bf16* row, int kbase, int lane) {
    const int ko = kbase + ((lane >> 4) << 4);
    FragU u;
    u.h[0] = *(const v8bf*)(row + ko);
    u.h[1] = *(const v8bf*)(row + ko + 8);
    return u.v;
}

// B fragment, 16-bit B 32x16: lane half holds K in [h*16, h*16+16) of column (lane&15).
// Column n of B == row n of W (row-major, stride HID) since we compute h @ W^T.
static __device__ __forceinline__ v16bf load_frag_b(const bf16* wrow, int kbase, int lane) {
    const int ko = kbase + ((lane >> 4) << 4);
    FragU u;
    u.h[0] = *(const v8bf*)(wrow + ko);
    u.h[1] = *(const v8bf*)(wrow + ko + 8);
    return u.v;
}

// B fragment from f32 source with in-register RNE conversion (projection kernel).
static __device__ __forceinline__ v16bf load_frag_b_f32(const float* wrow, int kbase, int lane) {
    const int ko = kbase + ((lane >> 4) << 4);
    FragU u;
#pragma unroll
    for (int i = 0; i < 4; ++i) {
        float4 f = *(const float4*)(wrow + ko + i * 4);
        u.s[i * 4 + 0] = f32_to_bf16_rne(f.x);
        u.s[i * 4 + 1] = f32_to_bf16_rne(f.y);
        u.s[i * 4 + 2] = f32_to_bf16_rne(f.z);
        u.s[i * 4 + 3] = f32_to_bf16_rne(f.w);
    }
    return u.v;
}

// Device-wide barrier across the NWG persistent workgroups.
// counter = bar[0], monotone gate = bar[1] (zeroed before launch).
static __device__ __forceinline__ void grid_barrier(unsigned* bar, unsigned step) {
    __threadfence();          // publish this wave's global stores (agent scope)
    __syncthreads();
    if (threadIdx.x == 0) {
        unsigned prev = __hip_atomic_fetch_add(&bar[0], 1u, __ATOMIC_ACQ_REL,
                                               __HIP_MEMORY_SCOPE_AGENT);
        if (prev == NWG - 1) {
            __hip_atomic_store(&bar[0], 0u, __ATOMIC_RELAXED, __HIP_MEMORY_SCOPE_AGENT);
            __hip_atomic_store(&bar[1], step, __ATOMIC_RELEASE, __HIP_MEMORY_SCOPE_AGENT);
        } else {
            while (__hip_atomic_load(&bar[1], __ATOMIC_ACQUIRE,
                                     __HIP_MEMORY_SCOPE_AGENT) < step) {
                __builtin_amdgcn_s_sleep(2);
            }
        }
    }
    __syncthreads();
    __threadfence();          // acquire side: invalidate stale lines for all waves
}

__global__ void rnn_zero_ws(unsigned* p, int n) {
    int i = blockIdx.x * blockDim.x + threadIdx.x;
    if (i < n) p[i] = 0u;
}

// ------------------------- persistent recurrent kernel ----------------------
__global__ void __launch_bounds__(THREADS)
rnn_recurrent_kernel(const int*   __restrict__ x,      // [64 x 512]
                     const float* __restrict__ Wxh,    // [32000 x 1024]
                     const float* __restrict__ Whh_w,  // [1024 x 1024]
                     const float* __restrict__ Whh_b,  // [1024]
                     bf16* __restrict__ Wbf,           // [1024 x 1024] bf16 scratch
                     bf16* __restrict__ h0,            // [64 x 1024] bf16 (zeroed)
                     bf16* __restrict__ h1,            // [64 x 1024] bf16
                     unsigned* __restrict__ bar) {
    const int tid  = threadIdx.x;
    const int wg   = blockIdx.x;
    const int lane = tid & 31;
    const int wv   = tid >> 5;

    // Phase 0: convert Whh_w f32 -> bf16 (2 MB, stays L2-resident for 512 steps).
    for (int i = (wg * THREADS + tid) * 4; i < HID * HID; i += NWG * THREADS * 4) {
        float4 f = *(const float4*)(Whh_w + i);
        uint2 p;
        p.x = (unsigned)f32_to_bf16_rne(f.x) | ((unsigned)f32_to_bf16_rne(f.y) << 16);
        p.y = (unsigned)f32_to_bf16_rne(f.z) | ((unsigned)f32_to_bf16_rne(f.w) << 16);
        *(uint2*)((unsigned short*)Wbf + i) = p;
    }

    // Tile ownership: WG covers 64 cols; wave -> M16 x N32 (two 16x16 C tiles).
    const int mbase = (wv & 3) * 16;
    const int nsub0 = wg * 64 + (wv >> 2) * 32;
    const int mrow  = mbase + ((lane >> 4) << 3);   // C-layout M base for this lane half
    const int ncol0 = nsub0 + (lane & 15);
    const int ncol1 = ncol0 + 16;
    const float bias0 = Whh_b[ncol0];
    const float bias1 = Whh_b[ncol1];

    const bf16* brow0 = Wbf + (nsub0 + (lane & 15)) * HID;
    const bf16* brow1 = brow0 + 16 * HID;

    for (int t = 0; t < TSEQ; ++t) {
        grid_barrier(bar, (unsigned)(t + 1));  // t=0: wait for Wbf; else wait for h_{t-1}

        const bf16* hp = (t & 1) ? h1 : h0;
        bf16*       hn = (t & 1) ? h0 : h1;
        const bf16* arow = hp + (mbase + (lane & 15)) * HID;

        // Issue the embedding gather for this timestep up front: it does not
        // depend on the GEMM, so its latency hides under the 32-iteration
        // WMMA loop instead of serializing after it.
        int toks[8];
#pragma unroll
        for (int r = 0; r < 8; ++r) toks[r] = x[(mrow + r) * TSEQ + t];
        float e0v[8], e1v[8];
#pragma unroll
        for (int r = 0; r < 8; ++r) {
            const float* erow = Wxh + (long)toks[r] * HID;
            e0v[r] = erow[ncol0];
            e1v[r] = erow[ncol1];
        }

        v8f c0 = {}, c1 = {};
#pragma unroll 4
        for (int kb = 0; kb < HID; kb += 32) {
            v16bf a  = load_frag_a(arow,  kb, lane);
            v16bf b0 = load_frag_b(brow0, kb, lane);
            v16bf b1 = load_frag_b(brow1, kb, lane);
            c0 = __builtin_amdgcn_wmma_f32_16x16x32_bf16(false, a, false, b0,
                                                         (short)0, c0, false, false);
            c1 = __builtin_amdgcn_wmma_f32_16x16x32_bf16(false, a, false, b1,
                                                         (short)0, c1, false, false);
        }

        // Fused epilogue: + embedding + bias, branch-free tanh, bf16 store of h_t.
#pragma unroll
        for (int r = 0; r < 8; ++r) {
            const int m = mrow + r;
            const float v0 = fast_tanh(c0[r] + e0v[r] + bias0);
            const float v1 = fast_tanh(c1[r] + e1v[r] + bias1);
            ((unsigned short*)hn)[m * HID + ncol0] = f32_to_bf16_rne(v0);
            ((unsigned short*)hn)[m * HID + ncol1] = f32_to_bf16_rne(v1);
        }
    }
}

// ------------------------- output projection kernel -------------------------
// 64 x 32000 = h @ Why_w^T + b. Memory bound (128 MB of Why_w); convert B tiles
// to bf16 in-register and use WMMA. WG covers 128 cols; wave -> M16 x N64.
__global__ void __launch_bounds__(256)
rnn_output_kernel(const bf16*  __restrict__ h,      // [64 x 1024] bf16 (final state)
                  const float* __restrict__ Why_w,  // [32000 x 1024]
                  const float* __restrict__ Why_b,  // [32000]
                  float*       __restrict__ out) {  // [64 x 32000]
    const int tid  = threadIdx.x;
    const int lane = tid & 31;
    const int wv   = tid >> 5;
    const int mbase = (wv & 3) * 16;
    const int nbase = blockIdx.x * 128 + (wv >> 2) * 64;

    const bf16* arow = h + (mbase + (lane & 15)) * HID;
    const float* brow[4];
#pragma unroll
    for (int s = 0; s < 4; ++s)
        brow[s] = Why_w + (long)(nbase + s * 16 + (lane & 15)) * HID;

    v8f c[4] = {};
    for (int kb = 0; kb < HID; kb += 32) {
        v16bf a = load_frag_a(arow, kb, lane);
#pragma unroll
        for (int s = 0; s < 4; ++s) {
            v16bf b = load_frag_b_f32(brow[s], kb, lane);
            c[s] = __builtin_amdgcn_wmma_f32_16x16x32_bf16(false, a, false, b,
                                                           (short)0, c[s], false, false);
        }
    }

    const int mrow = mbase + ((lane >> 4) << 3);
#pragma unroll
    for (int s = 0; s < 4; ++s) {
        const int n = nbase + s * 16 + (lane & 15);
        const float bias = Why_b[n];
#pragma unroll
        for (int r = 0; r < 8; ++r) {
            out[(long)(mrow + r) * NOUT + n] = c[s][r] + bias;
        }
    }
}

// ---------------------------------------------------------------------------
extern "C" void kernel_launch(void* const* d_in, const int* in_sizes, int n_in,
                              void* d_out, int out_size, void* d_ws, size_t ws_size,
                              hipStream_t stream) {
    const int*   x     = (const int*)d_in[0];
    const float* Wxh   = (const float*)d_in[1];
    const float* Whh_w = (const float*)d_in[2];
    const float* Whh_b = (const float*)d_in[3];
    const float* Why_w = (const float*)d_in[4];
    const float* Why_b = (const float*)d_in[5];
    float* out = (float*)d_out;

    // Workspace layout (≈2.3 MB): [bar 256B][h0 128KB][h1 128KB][Wbf 2MB]
    char* ws = (char*)d_ws;
    unsigned* bar = (unsigned*)ws;
    bf16* h0  = (bf16*)(ws + 256);
    bf16* h1  = (bf16*)(ws + 256 + BATCH * HID * sizeof(bf16));
    bf16* Wbf = (bf16*)(ws + 256 + 2 * BATCH * HID * sizeof(bf16));

    // Zero barrier words + h0 (initial hidden state). Contiguous region.
    const int zero_words = (256 + BATCH * HID * (int)sizeof(bf16)) / 4;
    rnn_zero_ws<<<(zero_words + 255) / 256, 256, 0, stream>>>((unsigned*)ws, zero_words);

    // One persistent kernel runs all 512 timesteps (16 WGs, device barrier between steps).
    rnn_recurrent_kernel<<<NWG, THREADS, 0, stream>>>(x, Wxh, Whh_w, Whh_b,
                                                      Wbf, h0, h1, bar);

    // TSEQ is even -> final hidden state ends up in h0.
    rnn_output_kernel<<<NOUT / 128, 256, 0, stream>>>(h0, Why_w, Why_b, out);
}
